// GraphAttentionLayer_10282151707284
// MI455X (gfx1250) — matrix operand order repaired and checked by reading, not compiled, not run on hardware
//
#include <hip/hip_runtime.h>
#include <hip/hip_bf16.h>

// GAT layer: N=2048 nodes, IN_F=512, 8 heads x 64 hidden.
// Inputs (setup_inputs order): h f32[2048*512], adj bool[2048*2048*1] (1 byte/elem),
// docnum i32, secnum i32, W_lin f32[512*512], attn_w f32[128].
// Output: f32[2048*512].

#define N_NODES 2048
#define IN_F    512
#define N_HEADS 8
#define N_HID   64
#define OUT_F   512            // N_HEADS * N_HID
#define NEG_SLOPE 0.2f
#define MASK_VAL (-1000000000.0f)

typedef __attribute__((ext_vector_type(16))) __bf16 v16bf;
typedef __attribute__((ext_vector_type(8)))  float  v8f;
typedef __attribute__((ext_vector_type(4)))  int    v4i;

// Async global->LDS staging (gfx1250): guarded so either toolchain compiles.
#if defined(__has_builtin)
#if __has_builtin(__builtin_amdgcn_global_load_async_to_lds_b128) && \
    __has_builtin(__builtin_amdgcn_s_wait_asynccnt)
#define HAVE_ASYNC_LDS 1
#endif
#endif
#ifndef HAVE_ASYNC_LDS
#define HAVE_ASYNC_LDS 0
#endif

#if HAVE_ASYNC_LDS
// Param 0 type probe-confirmed: 'int __vector(4) __device__ *' (global int4*).
typedef __attribute__((address_space(1))) v4i* gv4_t;
typedef __attribute__((address_space(3))) v4i* lv4_t;
__device__ inline void async_copy_b128(const void* gsrc, void* ldst) {
    __builtin_amdgcn_global_load_async_to_lds_b128((gv4_t)gsrc, (lv4_t)ldst, 0, 0);
}
__device__ inline void async_wait_all() { __builtin_amdgcn_s_wait_asynccnt(0); }
#endif

// Stage `nbytes/16` b128 chunks from global to LDS (async when available).
__device__ inline void stage_to_lds(const __bf16* gsrc, __bf16* ldst, int n_b128) {
#if HAVE_ASYNC_LDS
    #pragma unroll
    for (int c = 0; c < 4; ++c)
        if (c < n_b128) async_copy_b128(gsrc + c * 8, ldst + c * 8);
#else
    const uint4* s = (const uint4*)gsrc;
    uint4* d = (uint4*)ldst;
    #pragma unroll
    for (int c = 0; c < 4; ++c)
        if (c < n_b128) d[c] = s[c];
#endif
}

// ---------------------------------------------------------------- converts
__global__ __launch_bounds__(256)
void cvt_bf16_kernel(const float* __restrict__ src, __bf16* __restrict__ dst, int n) {
    int i = blockIdx.x * blockDim.x + threadIdx.x;
    if (i < n) dst[i] = (__bf16)src[i];
}

// W [512 k][512 n] f32 -> WbT [512 n][512 k] bf16 (one-time 1MB transpose, L2-resident)
__global__ __launch_bounds__(256)
void cvt_transpose_kernel(const float* __restrict__ W, __bf16* __restrict__ WbT) {
    int id = blockIdx.x * 256 + threadIdx.x;
    int k = id >> 9, n = id & 511;
    WbT[(size_t)n * IN_F + k] = (__bf16)W[id];
}

// ---------------------------------------------------------------- fragment helpers
// A fragment (16x32 bf16, M x K), contiguous k order (same as B; not the ISA table order):
//   lane l: M = l&15 ; holds K = kofs + (l>>4)*16 + {0..15}
//   -> one contiguous 32B run per lane => 2x ds_load_b128
__device__ inline v16bf load_A_frag(const __bf16* As, int ld, int kofs, int lane) {
    const __bf16* rowp = As + (lane & 15) * ld + kofs + ((lane >> 4) * 16);
    v16bf a;
#pragma unroll
    for (int e = 0; e < 16; ++e) { a[e] = rowp[e]; }
    return a;
}
// B fragment (32x16 bf16, K x N) read from an N-MAJOR tile Bt[n][k]:
//   lane l: N = ncol0 + (l&15) ; K = kofs + (l>>4)*16 + {0..15}
//   -> one contiguous 32B run per lane => 2x ds_load_b128, no packing movs
__device__ inline v16bf load_Bt_frag(const __bf16* Bts, int ld, int ncol0, int kofs, int lane) {
    const __bf16* rowp = Bts + (size_t)(ncol0 + (lane & 15)) * ld + kofs + (lane >> 4) * 16;
    v16bf b;
#pragma unroll
    for (int e = 0; e < 16; ++e) b[e] = rowp[e];
    return b;
}
// C/D f32 16x16: VGPR r, lane l -> M = 8*(l>>4) + r, N = l&15

// ---------------------------------------------------------------- GEMM1: g = h @ W
// block = 128 threads (4 waves). Block tile: 32 rows x 64 cols, K-chunk 64.
// Wave w owns cols w*16..+16; 2 M-tiles x 2 K-steps = 4 WMMA per stage, B reused.
// Tile staging uses the async global->LDS engine (no VGPR round trip).
__global__ __launch_bounds__(128)
void gemm_g_kernel(const __bf16* __restrict__ hb, const __bf16* __restrict__ WbT,
                   float* __restrict__ g, __bf16* __restrict__ gbT) {
    __shared__ __attribute__((aligned(16))) __bf16 As[32 * 64];   // [m][k] 4KB
    __shared__ __attribute__((aligned(16))) __bf16 Bts[64 * 64];  // [n][k] 8KB
    const int row0 = blockIdx.x * 32;
    const int col0 = blockIdx.y * 64;
    const int t = threadIdx.x, w = t >> 5, l = t & 31;

    v8f acc0 = {}, acc1 = {};
    for (int k0 = 0; k0 < IN_F; k0 += 64) {
        { // stage A 32x64 (each thread: 16 bf16 = 32B, coalesced)
            int r = t >> 2, kseg = (t & 3) * 16;
            stage_to_lds(hb + (size_t)(row0 + r) * IN_F + k0 + kseg,
                         &As[r * 64 + kseg], 2);
        }
        { // stage Bt 64n x 64k from pre-transposed WbT (each thread: 32 bf16 = 64B)
            int n = t >> 1, kseg = (t & 1) * 32;
            stage_to_lds(WbT + (size_t)(col0 + n) * IN_F + k0 + kseg,
                         &Bts[n * 64 + kseg], 4);
        }
#if HAVE_ASYNC_LDS
        async_wait_all();   // drain this wave's ASYNCcnt before barrier signal
#endif
        __syncthreads();
#pragma unroll
        for (int ks = 0; ks < 64; ks += 32) {
            v16bf b  = load_Bt_frag(Bts, 64, w * 16, ks, l);
            v16bf a0 = load_A_frag(As,       64, ks, l);
            v16bf a1 = load_A_frag(As + 16 * 64, 64, ks, l);
            acc0 = __builtin_amdgcn_wmma_f32_16x16x32_bf16(false, a0, false, b,
                                                           (short)0, acc0, false, false);
            acc1 = __builtin_amdgcn_wmma_f32_16x16x32_bf16(false, a1, false, b,
                                                           (short)0, acc1, false, false);
        }
        __syncthreads();
    }
    { // store: g row-major f32; gbT column(feature)-major bf16 (contiguous 16B per lane)
        int n = col0 + w * 16 + (l & 15);
#pragma unroll
        for (int mt = 0; mt < 2; ++mt) {
            v8f acc = mt ? acc1 : acc0;
            int rbase = row0 + mt * 16 + 8 * (l >> 4);
            __bf16 tmp[8];
#pragma unroll
            for (int r = 0; r < 8; ++r) {
                float v = acc[r];
                g[(size_t)(rbase + r) * OUT_F + n] = v;
                tmp[r] = (__bf16)v;
            }
            *(uint4*)(&gbT[(size_t)n * N_NODES + rbase]) = *(const uint4*)tmp;
        }
    }
}

// ---------------------------------------------------------------- scores s_i, s_j
__global__ __launch_bounds__(256)
void scores_kernel(const float* __restrict__ g, const float* __restrict__ attn_w,
                   float* __restrict__ s_i, float* __restrict__ s_j) {
    int id = blockIdx.x * blockDim.x + threadIdx.x;
    if (id >= N_NODES * N_HEADS) return;
    int n = id >> 3, hd = id & 7;
    const float* gp = g + (size_t)n * OUT_F + hd * N_HID;
    float si = 0.f, sj = 0.f;
#pragma unroll 8
    for (int f = 0; f < N_HID; ++f) {
        float v = gp[f];
        si += v * attn_w[f];
        sj += v * attn_w[N_HID + f];
    }
    s_i[id] = si;
    s_j[id] = sj;
}

// ---------------------------------------------------------------- fused softmax + aggregation
// grid = (128 row-tiles, 8 heads); block = 128 threads (4 waves).
// Block computes out[i0:i0+16, head*64:(head+1)*64] without materializing attention.
// g-tile staging via async global->LDS, overlapped with the softmax VALU work.
__global__ __launch_bounds__(128)
void attn_kernel(const float* __restrict__ s_i, const float* __restrict__ s_j,
                 const unsigned char* __restrict__ adj, const __bf16* __restrict__ gbT,
                 float* __restrict__ out) {
    __shared__ float sI[16];
    __shared__ float pm[16][8];
    __shared__ float pl[16][8];
    __shared__ float m_s[16];
    __shared__ float li_s[16];
    __shared__ __attribute__((aligned(16))) __bf16 p_tile[16 * 64]; // [i][j] 2KB
    __shared__ __attribute__((aligned(16))) __bf16 Bts[64 * 64];    // [f][j] 8KB

    const int i0 = blockIdx.x * 16;
    const int head = blockIdx.y;
    const int t = threadIdx.x, w = t >> 5, l = t & 31;

    if (t < 16) sI[t] = s_i[(size_t)(i0 + t) * N_HEADS + head];
    __syncthreads();

    // ---- pass 1: per-row online max + exp-sum over all j (8 threads per row)
    {
        int r = t >> 3, slot = t & 7;
        float si = sI[r];
        const unsigned char* arow = adj + (size_t)(i0 + r) * N_NODES;
        float m = -3.0e38f, lsum = 0.f;
        for (int j = slot; j < N_NODES; j += 8) {
            __builtin_prefetch(arow + j + 512, 0, 1); // stream adj via global_prefetch
            float e = si + s_j[(size_t)j * N_HEADS + head];
            e = (e > 0.f) ? e : NEG_SLOPE * e;
            if (!arow[j]) e = MASK_VAL;
            float mn = fmaxf(m, e);
            lsum = lsum * __expf(m - mn) + __expf(e - mn);
            m = mn;
        }
        pm[r][slot] = m; pl[r][slot] = lsum;
    }
    __syncthreads();
    if (t < 16) {
        float m = pm[t][0], lsum = pl[t][0];
#pragma unroll
        for (int s2 = 1; s2 < 8; ++s2) {
            float m2 = pm[t][s2], l2 = pl[t][s2];
            float mn = fmaxf(m, m2);
            lsum = lsum * __expf(m - mn) + l2 * __expf(m2 - mn);
            m = mn;
        }
        m_s[t] = m;
        li_s[t] = 1.0f / lsum;
    }
    __syncthreads();

    // ---- pass 2: stream 64-wide j chunks: bf16 p tile -> WMMA against g tile
    v8f acc = {};
    for (int j0 = 0; j0 < N_NODES; j0 += 64) {
        { // stage Bt 64f x 64j from gbT (each thread: 32 bf16 = 64B, coalesced);
          // async DMA overlaps with the probability-tile math below.
            int f = t >> 1, jseg = (t & 1) * 32;
            stage_to_lds(gbT + (size_t)(head * N_HID + f) * N_NODES + j0 + jseg,
                         &Bts[f * 64 + jseg], 4);
        }
        // compute probability tile (16 x 64) in bf16 (independent of g staging)
#pragma unroll
        for (int q = 0; q < 8; ++q) {
            int idx = t + q * 128;
            int r = idx >> 6, c = idx & 63;
            int j = j0 + c;
            float e = sI[r] + s_j[(size_t)j * N_HEADS + head];
            e = (e > 0.f) ? e : NEG_SLOPE * e;
            if (!adj[(size_t)(i0 + r) * N_NODES + j]) e = MASK_VAL;
            float p = __expf(e - m_s[r]) * li_s[r];
            p_tile[r * 64 + c] = (__bf16)p;
        }
#if HAVE_ASYNC_LDS
        async_wait_all();   // drain this wave's ASYNCcnt before barrier signal
#endif
        __syncthreads();
#pragma unroll
        for (int ks = 0; ks < 64; ks += 32) {
            v16bf a = load_A_frag(p_tile, 64, ks, l);
            v16bf b = load_Bt_frag(Bts, 64, w * 16, ks, l);
            acc = __builtin_amdgcn_wmma_f32_16x16x32_bf16(false, a, false, b,
                                                          (short)0, acc, false, false);
        }
        __syncthreads();
    }
    {
        int n = head * N_HID + w * 16 + (l & 15);
        int rbase = i0 + 8 * (l >> 4);
#pragma unroll
        for (int r = 0; r < 8; ++r)
            out[(size_t)(rbase + r) * OUT_F + n] = acc[r];
    }
}

// ---------------------------------------------------------------- launch
extern "C" void kernel_launch(void* const* d_in, const int* in_sizes, int n_in,
                              void* d_out, int out_size, void* d_ws, size_t ws_size,
                              hipStream_t stream) {
    const float*         h      = (const float*)d_in[0];
    const unsigned char* adj    = (const unsigned char*)d_in[1]; // jnp bool = 1 byte/elem
    const float*         W      = (const float*)d_in[4];
    const float*         attn_w = (const float*)d_in[5];
    float*               out    = (float*)d_out;

    // workspace layout
    char* ws = (char*)d_ws;
    __bf16* hb  = (__bf16*)(ws);                                              // 2 MB
    __bf16* WbT = (__bf16*)(ws + (size_t)(2u << 20));                         // 0.5 MB  [n][k]
    float*  g   = (float*) (ws + (size_t)(2u << 20) + (512u << 10));          // 4 MB    [n][f]
    __bf16* gbT = (__bf16*)(ws + (size_t)(2u << 20) + (512u << 10) + (4u << 20)); // 2 MB [f][j]
    float*  s_i = (float*) (ws + (size_t)(2u << 20) + (512u << 10) + (6u << 20)); // 64 KB
    float*  s_j = s_i + N_NODES * N_HEADS;                                    // 64 KB

    cvt_bf16_kernel<<<(N_NODES * IN_F) / 256, 256, 0, stream>>>(h, hb, N_NODES * IN_F);
    cvt_transpose_kernel<<<(IN_F * OUT_F) / 256, 256, 0, stream>>>(W, WbT);

    gemm_g_kernel<<<dim3(N_NODES / 32, OUT_F / 64), 128, 0, stream>>>(hb, WbT, g, gbT);

    scores_kernel<<<(N_NODES * N_HEADS) / 256, 256, 0, stream>>>(g, attn_w, s_i, s_j);

    attn_kernel<<<dim3(N_NODES / 16, N_HEADS), 128, 0, stream>>>(s_i, s_j, adj, gbT, out);
}
